// CustomSTFT_19301583028452
// MI455X (gfx1250) — matrix operand order repaired and checked by reading, not compiled, not run on hardware
//
#include <hip/hip_runtime.h>
#include <hip/hip_bf16.h>
#include <math.h>

typedef __attribute__((ext_vector_type(16))) _Float16 v16h;
typedef __attribute__((ext_vector_type(8)))  float    v8f;

#define NFFT   800
#define HOPS   200
#define FREQ   401
#define BATCH  32
#define TLEN   480000
#define NFR    2401                  // frames per batch
#define MROWS  (BATCH * NFR)         // 76832 total frames
#define NCOLS  832                   // padded spectrum width (416 real + 416 imag)
#define HALFC  416
#define PADV   400                   // N_FFT/2 edge pad
#define XROW   480832                // padded signal row: 400 + 480000 + 400 + 32 tail
#define KTOT   832                   // unified GEMM depth (rows >= 800 have zero weights)
#define BM     128                   // block M (4 waves x 32 rows)

// ---------------------------------------------------------------------------
// Edge-padded f16 signal: xp[b][t] = x[b][clamp(t-400, 0, TLEN-1)]
// ---------------------------------------------------------------------------
__global__ void prep_pad_signal(const float* __restrict__ x, _Float16* __restrict__ xp) {
  long idx = (long)blockIdx.x * 256 + threadIdx.x;
  if (idx >= (long)BATCH * XROW) return;
  int b = (int)(idx / XROW);
  int t = (int)(idx % XROW);
  int s = t - PADV;
  s = s < 0 ? 0 : (s >= TLEN ? TLEN - 1 : s);
  xp[idx] = (_Float16)x[(long)b * TLEN + s];
}

// ---------------------------------------------------------------------------
// Weight prep: pack DFT filterbank into [832 x 832] f16 B matrices.
// ---------------------------------------------------------------------------
__global__ void prep_fwd_weights(const float* __restrict__ wfr,
                                 const float* __restrict__ wfi,
                                 _Float16* __restrict__ Wf1) {
  int idx = blockIdx.x * 256 + threadIdx.x;
  if (idx >= KTOT * NCOLS) return;
  int n = idx / NCOLS, c = idx % NCOLS;
  float v = 0.0f;
  if (n < NFFT) {
    if (c < FREQ)                          v = wfr[c * NFFT + n];
    else if (c >= HALFC && c < HALFC+FREQ) v = wfi[(c - HALFC) * NFFT + n];
  }
  Wf1[idx] = (_Float16)v;
}

__global__ void prep_bwd_weights(const float* __restrict__ wbr,
                                 const float* __restrict__ wbi,
                                 _Float16* __restrict__ Wf2) {
  int idx = blockIdx.x * 256 + threadIdx.x;
  if (idx >= KTOT * NCOLS) return;
  int k = idx / NCOLS, n = idx % NCOLS;
  float v = 0.0f;
  if (n < NFFT) {
    if (k < FREQ)                          v =  wbr[k * NFFT + n];
    else if (k >= HALFC && k < HALFC+FREQ) v = -wbi[(k - HALFC) * NFFT + n];
  }
  Wf2[idx] = (_Float16)v;
}

// ---------------------------------------------------------------------------
// LDS tile copies (all addresses 16B/8B aligned by construction)
// ---------------------------------------------------------------------------
__device__ __forceinline__ void cpA64(_Float16* dst, const _Float16* src) {
  const uint4* s = (const uint4*)src;
  uint4* d = (uint4*)dst;
#pragma unroll
  for (int j = 0; j < 8; ++j) d[j] = s[j];   // 64 halves (one full A row)
}
__device__ __forceinline__ void cpB32(_Float16* dst, const _Float16* src) {
  const uint2* s = (const uint2*)src;
  uint2* d = (uint2*)dst;
#pragma unroll
  for (int j = 0; j < 8; ++j) d[j] = s[j];   // 32 halves
}

// ---------------------------------------------------------------------------
// WMMA core: 64-deep K block for a 128Mx64N tile. 4 waves; each wave owns
// 32Mx64N -> 8 accumulators, 16 WMMAs. All fragments are preloaded so LDS
// latency is covered by one wait, then WMMAs issue back-to-back.
// A frag (16x32): lane = M, K striped over VGPR pairs + lane halves.
// B frag (32x16): lane = N, K striped over VGPR pairs + lane halves. (CDNA5 ISA sec 7.12.2)
// ---------------------------------------------------------------------------
__device__ __forceinline__
void compute_k64(const _Float16 (&As)[BM][72], const _Float16 (&Bs)[64][68],
                 v8f (&acc)[2][4], int lane, int wave) {
  const int ml = lane & 15;
  const int ah = (lane >> 4) * 8;
  v16h af[2][2];   // [ks][mt]
  v16h bf[2][4];   // [ks][nt]
#pragma unroll
  for (int ks2 = 0; ks2 < 2; ++ks2) {
#pragma unroll
    for (int mt = 0; mt < 2; ++mt) {
      const int am = wave * 32 + mt * 16 + ml;
#pragma unroll
      for (int v = 0; v < 8; ++v) {
        int kb = ks2 * 32 + (v >> 2) * 16 + ah + (v & 3) * 2;
        af[ks2][mt][2 * v]     = As[am][kb];
        af[ks2][mt][2 * v + 1] = As[am][kb + 1];
      }
    }
#pragma unroll
    for (int nt = 0; nt < 4; ++nt) {
#pragma unroll
      for (int v = 0; v < 8; ++v) {
        int kb = ks2 * 32 + (v >> 2) * 16 + ah + (v & 3) * 2;
        bf[ks2][nt][2 * v]     = Bs[kb][nt * 16 + ml];
        bf[ks2][nt][2 * v + 1] = Bs[kb + 1][nt * 16 + ml];
      }
    }
  }
#pragma unroll
  for (int ks2 = 0; ks2 < 2; ++ks2)
#pragma unroll
    for (int mt = 0; mt < 2; ++mt)
#pragma unroll
      for (int nt = 0; nt < 4; ++nt)
        acc[mt][nt] = __builtin_amdgcn_wmma_f32_16x16x32_f16(
            false, af[ks2][mt], false, bf[ks2][nt], (short)0, acc[mt][nt], false, false);
}

__device__ __forceinline__
void store_tile(v8f (&acc)[2][4], _Float16* __restrict__ dst,
                int g0, int n0, int lane, int wave) {
  const int cn = lane & 15;
  const int cm = (lane >> 4) * 8;
#pragma unroll
  for (int mt = 0; mt < 2; ++mt)
#pragma unroll
    for (int nt = 0; nt < 4; ++nt)
#pragma unroll
      for (int r = 0; r < 8; ++r) {
        int row = g0 + wave * 32 + mt * 16 + cm + r;
        if (row < MROWS)
          dst[(long)row * NCOLS + n0 + nt * 16 + cn] = (_Float16)acc[mt][nt][r];
      }
}

// ---------------------------------------------------------------------------
// Forward: frames (contiguous slices of padded signal) @ Wf1 -> spec f16
// ---------------------------------------------------------------------------
__global__ __launch_bounds__(128)
void stft_fwd_gemm(const _Float16* __restrict__ xp, const _Float16* __restrict__ Wf1,
                   _Float16* __restrict__ spec) {
  __shared__ _Float16 As[2][BM][72];
  __shared__ _Float16 Bs[2][64][68];
  const int tid  = threadIdx.x;
  const int lane = tid & 31, wave = tid >> 5;
  const int g0 = blockIdx.x * BM;
  const int n0 = blockIdx.y * 64;

  v8f acc[2][4];
#pragma unroll
  for (int mt = 0; mt < 2; ++mt)
#pragma unroll
    for (int nt = 0; nt < 4; ++nt)
#pragma unroll
      for (int r = 0; r < 8; ++r) acc[mt][nt][r] = 0.0f;

  // A loader: one full row (64 halves) per thread; rows past MROWS re-read last row
  int g = g0 + tid; if (g >= MROWS) g = MROWS - 1;
  const int bb = g / NFR, ff = g % NFR;
  const _Float16* asrc = xp + (long)bb * XROW + ff * HOPS;
  // B loader: thread -> (row, 32-wide n segment)
  const int br   = tid >> 1;
  const int bseg = (tid & 1) * 32;
  const _Float16* bsrc = Wf1 + (long)br * NCOLS + n0 + bseg;

  cpA64(&As[0][tid][0], asrc);
  cpB32(&Bs[0][br][bseg], bsrc);

  int buf = 0;
  for (int k0 = 0; k0 < KTOT - 64; k0 += 64) {
    __syncthreads();
    __builtin_prefetch(asrc + k0 + 128, 0, 1);            // global_prefetch_b8
    cpA64(&As[buf ^ 1][tid][0], asrc + k0 + 64);
    cpB32(&Bs[buf ^ 1][br][bseg], bsrc + (long)(k0 + 64) * NCOLS);
    compute_k64(As[buf], Bs[buf], acc, lane, wave);
    buf ^= 1;
  }
  __syncthreads();
  compute_k64(As[buf], Bs[buf], acc, lane, wave);

  store_tile(acc, spec, g0, n0, lane, wave);
}

// ---------------------------------------------------------------------------
// Backward: spec @ Wf2 -> y_frames f16 [MROWS x NCOLS]
// ---------------------------------------------------------------------------
__global__ __launch_bounds__(128)
void stft_bwd_gemm(const _Float16* __restrict__ spec, const _Float16* __restrict__ Wf2,
                   _Float16* __restrict__ y) {
  __shared__ _Float16 As[2][BM][72];
  __shared__ _Float16 Bs[2][64][68];
  const int tid  = threadIdx.x;
  const int lane = tid & 31, wave = tid >> 5;
  const int g0 = blockIdx.x * BM;
  const int n0 = blockIdx.y * 64;

  v8f acc[2][4];
#pragma unroll
  for (int mt = 0; mt < 2; ++mt)
#pragma unroll
    for (int nt = 0; nt < 4; ++nt)
#pragma unroll
      for (int r = 0; r < 8; ++r) acc[mt][nt][r] = 0.0f;

  int g = g0 + tid; if (g >= MROWS) g = MROWS - 1;
  const _Float16* asrc = spec + (long)g * NCOLS;
  const int br   = tid >> 1;
  const int bseg = (tid & 1) * 32;
  const _Float16* bsrc = Wf2 + (long)br * NCOLS + n0 + bseg;

  cpA64(&As[0][tid][0], asrc);
  cpB32(&Bs[0][br][bseg], bsrc);

  int buf = 0;
  for (int k0 = 0; k0 < KTOT - 64; k0 += 64) {
    __syncthreads();
    cpA64(&As[buf ^ 1][tid][0], asrc + k0 + 64);
    cpB32(&Bs[buf ^ 1][br][bseg], bsrc + (long)(k0 + 64) * NCOLS);
    compute_k64(As[buf], Bs[buf], acc, lane, wave);
    buf ^= 1;
  }
  __syncthreads();
  compute_k64(As[buf], Bs[buf], acc, lane, wave);

  store_tile(acc, y, g0, n0, lane, wave);
}

// ---------------------------------------------------------------------------
// Pointwise magnitude/phase roundtrip (in place on f16 spectrum)
// ---------------------------------------------------------------------------
__global__ void mag_phase(_Float16* __restrict__ spec) {
  long idx = (long)blockIdx.x * 256 + threadIdx.x;
  if (idx >= (long)MROWS * HALFC) return;
  long gidx = idx / HALFC;
  int  k    = (int)(idx % HALFC);
  _Float16* pr = spec + gidx * NCOLS + k;
  _Float16* pi = pr + HALFC;
  float r = (float)*pr, i = (float)*pi;
  float mag = sqrtf(r * r + i * i + 1e-14f);
  float ph  = atan2f(i, r);
  if (i == 0.0f && r < 0.0f) ph = 3.14159265358979323846f;
  *pr = (_Float16)(mag * cosf(ph));
  *pi = (_Float16)(mag * sinf(ph));
}

// ---------------------------------------------------------------------------
// Overlap-add as a deterministic gather (<=4 frames per sample), no atomics.
// ---------------------------------------------------------------------------
__global__ void overlap_add(const _Float16* __restrict__ y, float* __restrict__ out) {
  long idx = (long)blockIdx.x * 256 + threadIdx.x;
  if (idx >= (long)BATCH * TLEN) return;
  int b  = (int)(idx / TLEN);
  int t  = (int)(idx % TLEN);
  int tp = t + PADV;
  int fhi = tp / HOPS;                   if (fhi > NFR - 1) fhi = NFR - 1;
  int flo = (tp - (NFFT - HOPS)) / HOPS; if (flo < 0) flo = 0;
  float s = 0.0f;
  for (int f = flo; f <= fhi; ++f) {
    int col = tp - f * HOPS;             // in [0, 800)
    s += (float)y[(long)(b * NFR + f) * NCOLS + col];
  }
  out[idx] = s;
}

// ---------------------------------------------------------------------------
extern "C" void kernel_launch(void* const* d_in, const int* in_sizes, int n_in,
                              void* d_out, int out_size, void* d_ws, size_t ws_size,
                              hipStream_t stream) {
  (void)in_sizes; (void)n_in; (void)out_size; (void)ws_size;
  const float* x   = (const float*)d_in[0];
  const float* wfr = (const float*)d_in[1];
  const float* wfi = (const float*)d_in[2];
  const float* wbr = (const float*)d_in[3];
  const float* wbi = (const float*)d_in[4];

  char* ws = (char*)d_ws;
  const size_t SZ_XP   = (size_t)BATCH * XROW * sizeof(_Float16);    // 30,773,248
  const size_t SZ_WF   = (size_t)KTOT * NCOLS * sizeof(_Float16);    // 1,384,448
  const size_t SZ_SPEC = (size_t)MROWS * NCOLS * sizeof(_Float16);   // 127,848,448
  _Float16* xp   = (_Float16*)(ws);
  _Float16* Wf1  = (_Float16*)(ws + SZ_XP);
  _Float16* Wf2  = (_Float16*)(ws + SZ_XP + SZ_WF);
  _Float16* spec = (_Float16*)(ws + SZ_XP + 2 * SZ_WF);
  _Float16* yb   = (_Float16*)(ws + SZ_XP + 2 * SZ_WF + SZ_SPEC);
  float* out = (float*)d_out;

  long np = (long)BATCH * XROW;
  prep_pad_signal<<<(unsigned)((np + 255) / 256), 256, 0, stream>>>(x, xp);
  prep_fwd_weights<<<(KTOT * NCOLS + 255) / 256, 256, 0, stream>>>(wfr, wfi, Wf1);
  prep_bwd_weights<<<(KTOT * NCOLS + 255) / 256, 256, 0, stream>>>(wbr, wbi, Wf2);

  dim3 gg((MROWS + BM - 1) / BM, NCOLS / 64);   // (601, 13)
  stft_fwd_gemm<<<gg, 128, 0, stream>>>(xp, Wf1, spec);

  long pw = (long)MROWS * HALFC;
  mag_phase<<<(unsigned)((pw + 255) / 256), 256, 0, stream>>>(spec);

  stft_bwd_gemm<<<gg, 128, 0, stream>>>(spec, Wf2, yb);

  long ow = (long)BATCH * TLEN;
  overlap_add<<<(unsigned)((ow + 255) / 256), 256, 0, stream>>>(yb, out);
}
